// HyperGraphStructuralLayer_Knn_19825569038841
// MI455X (gfx1250) — matrix operand order repaired and checked by reading, not compiled, not run on hardware
//
#include <hip/hip_runtime.h>
#include <hip/hip_bf16.h>

// HyperGraphStructuralLayer_Knn for MI455X (gfx1250, wave32, WMMA)
// N=16384 nodes, HIDDEN=256, K=5 neighbors (4 after dropping self).
//
// Cost model: KNN Gram matrix = 137 GFLOP (f16 WMMA, X fits in L2 at 8MB);
// layer GEMMs ~2 GFLOP each; scatter/gather ~100MB passes vs 23.3 TB/s HBM.
// KNN selection is the VALU hot path -> early-skip min test per 8-candidate
// segment so the ~10-op insertion runs only when the tile can improve top-4.

typedef __attribute__((ext_vector_type(16))) _Float16 v16h;
typedef __attribute__((ext_vector_type(8)))  _Float16 v8h;
typedef __attribute__((ext_vector_type(8)))  float    v8f;
typedef __attribute__((ext_vector_type(4)))  float    v4f;

#define NN   16384
#define HID  256
#define KNBR 4

// ---------- helpers ----------

// A-fragment loader for V_WMMA_F32_16X16X32_F16.
// 16-bit A 16x32 layout: lane l holds row M=l%16; VGPRs0-3 = K (l<16 ? 0..7 : 8..15),
// VGPRs4-7 = same +16.  base = &row[chunk*32 + (lane>>4)*8]; elements 0..7 from base,
// elements 8..15 from base+16.
static __device__ __forceinline__ v16h load_a_frag(const _Float16* base) {
  v8h lo = *(const v8h*)(base);
  v8h hi = *(const v8h*)(base + 16);
  return __builtin_shufflevector(lo, hi, 0,1,2,3,4,5,6,7,8,9,10,11,12,13,14,15);
}

// Sorted (ascending) branch-light top-4 smallest insertion.
static __device__ __forceinline__ void top4_insert(float d, int id,
    float& t0, float& t1, float& t2, float& t3,
    int& i0, int& i1, int& i2, int& i3)
{
  if (d < t3) {
    bool c2 = d < t2, c1 = d < t1, c0 = d < t0;
    t3 = c2 ? t2 : d;             i3 = c2 ? i2 : id;
    t2 = c2 ? (c1 ? t1 : d) : t2; i2 = c2 ? (c1 ? i1 : id) : i2;
    t1 = c1 ? (c0 ? t0 : d) : t1; i1 = c1 ? (c0 ? i0 : id) : i1;
    t0 = c0 ? d : t0;             i0 = c0 ? id : i0;
  }
}

static __device__ __forceinline__ void atomic_add_f32(float* p, float v) {
  unsafeAtomicAdd(p, v);  // -> global_atomic_add_f32 (no CAS loop)
}

// ---------- tiny utility kernels ----------

__global__ void zero_b32_kernel(unsigned* __restrict__ p, int n) {
  for (int i = blockIdx.x * blockDim.x + threadIdx.x; i < n; i += gridDim.x * blockDim.x)
    p[i] = 0u;
}

__global__ void f32_to_f16_kernel(const float* __restrict__ src, _Float16* __restrict__ dst, int n) {
  for (int i = blockIdx.x * blockDim.x + threadIdx.x; i < n; i += gridDim.x * blockDim.x)
    dst[i] = (_Float16)src[i];
}

__global__ void rownorm_kernel(const float* __restrict__ x, float* __restrict__ sq) {
  __shared__ float red[HID];
  const int i = blockIdx.x, c = threadIdx.x;
  float v = x[(size_t)i * HID + c];
  red[c] = v * v;
  __syncthreads();
  for (int s = HID / 2; s > 0; s >>= 1) {
    if (c < s) red[c] += red[c + s];
    __syncthreads();
  }
  if (c == 0) sq[i] = red[0];
}

// ---------- KNN: tiled Gram via WMMA + streaming top-4 ----------
// One wave per 16-row tile. A-fragments (8 x v16h = 64 VGPRs) preloaded once,
// then sweep 1024 column tiles of 16 (B = X^T, so same row-major loads from X).
__global__ __launch_bounds__(32) void knn_kernel(const _Float16* __restrict__ xh,
                                                 const float* __restrict__ sq,
                                                 int* __restrict__ nbr,
                                                 int* __restrict__ deg)
{
  __shared__ __attribute__((aligned(32))) float tile[16 * 16];
  __shared__ float mv[32 * 4];
  __shared__ int   mi[32 * 4];

  const int lane = threadIdx.x;
  const int hf = lane >> 4;     // 0/1: lane half
  const int lr = lane & 15;
  const int rowBase = blockIdx.x * 16;
  const float FINF = __builtin_inff();

  v16h af[8];
  {
    const _Float16* rp = xh + (size_t)(rowBase + lr) * HID + hf * 16;
#pragma unroll
    for (int c = 0; c < 8; ++c) af[c] = *(const v16h*)(rp + c * 32);
  }

  float t0 = FINF, t1 = FINF, t2 = FINF, t3 = FINF;
  int   i0 = 0,    i1 = 0,    i2 = 0,    i3 = 0;

  for (int jt = 0; jt < NN / 16; ++jt) {
    const int colBase = jt * 16;
    v8f acc = {};
    // 16-bit B 32x16: lane l holds column N=l%16, K = (l>>4)*16 + e (e=0..15),
    // i.e. 16 contiguous halfs of X row (colBase + l%16).
    const _Float16* bp = xh + (size_t)(colBase + lr) * HID + hf * 16;
#pragma unroll
    for (int c = 0; c < 8; ++c) {
      v16h b = *(const v16h*)(bp + c * 32);
      acc = __builtin_amdgcn_wmma_f32_16x16x32_f16(false, af[c], false, b,
                                                   (short)0, acc, false, false);
    }
    // Relative distance key per row: sq[col] - 2*dot (row constant dropped).
    const float sqc = sq[colBase + lr];        // C element (v, lane): M=v+8*hf, N=lr
    float dv[8];
#pragma unroll
    for (int v = 0; v < 8; ++v)
      dv[v] = __builtin_fmaf(-2.0f, acc[v], sqc);
    if (jt == blockIdx.x) {                    // self only occurs in diagonal tile
#pragma unroll
      for (int v = 0; v < 8; ++v)
        if (lr == v + 8 * hf) dv[v] = FINF;
    }
#pragma unroll
    for (int v = 0; v < 8; ++v)
      tile[(v + 8 * hf) * 16 + lr] = dv[v];
    __syncthreads();

    // Row lr is scanned by lanes lr (cols 0..7) and lr+16 (cols 8..15).
    // Fast path: two b128 LDS loads + 8-way min; insert only if it can improve.
    const v4f q0 = *(const v4f*)&tile[lr * 16 + hf * 8];
    const v4f q1 = *(const v4f*)&tile[lr * 16 + hf * 8 + 4];
    float mn = fminf(fminf(fminf(q0[0], q0[1]), fminf(q0[2], q0[3])),
                     fminf(fminf(q1[0], q1[1]), fminf(q1[2], q1[3])));
    if (mn < t3) {
      const int base = colBase + hf * 8;
#pragma unroll
      for (int e = 0; e < 4; ++e)
        top4_insert(q0[e], base + e,     t0, t1, t2, t3, i0, i1, i2, i3);
#pragma unroll
      for (int e = 0; e < 4; ++e)
        top4_insert(q1[e], base + 4 + e, t0, t1, t2, t3, i0, i1, i2, i3);
    }
    __syncthreads();
  }

  // Merge the two per-row partial top-4 lists.
  mv[lane * 4 + 0] = t0; mv[lane * 4 + 1] = t1; mv[lane * 4 + 2] = t2; mv[lane * 4 + 3] = t3;
  mi[lane * 4 + 0] = i0; mi[lane * 4 + 1] = i1; mi[lane * 4 + 2] = i2; mi[lane * 4 + 3] = i3;
  __syncthreads();
  if (lane < 16) {
#pragma unroll
    for (int k = 0; k < 4; ++k)
      top4_insert(mv[(lane + 16) * 4 + k], mi[(lane + 16) * 4 + k],
                  t0, t1, t2, t3, i0, i1, i2, i3);
    const int row = rowBase + lane;
    nbr[row * 4 + 0] = i0; nbr[row * 4 + 1] = i1;
    nbr[row * 4 + 2] = i2; nbr[row * 4 + 3] = i3;
    atomicAdd(&deg[i0], 1); atomicAdd(&deg[i1], 1);
    atomicAdd(&deg[i2], 1); atomicAdd(&deg[i3], 1);
  }
}

// ---------- C[M x 256] = A[M x 256] @ B[256 x 256]^T via WMMA ----------
// B rows are the WMMA B-matrix columns -> same loader as A's data source.
__global__ __launch_bounds__(32) void gemm_kernel(const _Float16* __restrict__ Ah,
                                                  const _Float16* __restrict__ Bh,
                                                  float* __restrict__ C)
{
  const int lane = threadIdx.x;
  const int hf = lane >> 4, lr = lane & 15;
  const int rowBase = blockIdx.x * 16;

  v16h af[8];
  {
    const _Float16* rp = Ah + (size_t)(rowBase + lr) * HID + hf * 16;
#pragma unroll
    for (int c = 0; c < 8; ++c) af[c] = *(const v16h*)(rp + c * 32);
  }

  for (int ct = 0; ct < HID / 16; ++ct) {
    const int colBase = ct * 16;
    v8f acc = {};
    const _Float16* bp = Bh + (size_t)(colBase + lr) * HID + hf * 16;
#pragma unroll
    for (int c = 0; c < 8; ++c) {
      v16h b = *(const v16h*)(bp + c * 32);
      acc = __builtin_amdgcn_wmma_f32_16x16x32_f16(false, af[c], false, b,
                                                   (short)0, acc, false, false);
    }
#pragma unroll
    for (int v = 0; v < 8; ++v)
      C[(size_t)(rowBase + v + 8 * hf) * HID + colBase + lr] = acc[v];
  }
}

// ---------- hypergraph scatter / gather ----------

__global__ void scatter_kernel(const float* __restrict__ xt, const int* __restrict__ nbr,
                               float* __restrict__ ef)
{
  const int i = blockIdx.x, c = threadIdx.x;
  float v = xt[(size_t)i * HID + c];
#pragma unroll
  for (int k = 0; k < KNBR; ++k)
    atomic_add_f32(&ef[(size_t)nbr[i * KNBR + k] * HID + c], v);
}

// out[i] = prelu( 0.25 * sum_k Binv[nbr_k] * ef[nbr_k] + bias  (+ resid) )
__global__ void gather_kernel(const float* __restrict__ ef, const int* __restrict__ deg,
                              const int* __restrict__ nbr, const float* __restrict__ bias,
                              const float* __restrict__ resid, const float* __restrict__ prelu_a,
                              float* __restrict__ out)
{
  const int i = blockIdx.x, c = threadIdx.x;
  float s = 0.0f;
#pragma unroll
  for (int k = 0; k < KNBR; ++k) {
    const int e = nbr[i * KNBR + k];
    const int dg = deg[e];
    const float binv = dg > 0 ? 1.0f / (float)dg : 0.0f;
    s += binv * ef[(size_t)e * HID + c];
  }
  float h = 0.25f * s + bias[c];
  if (resid) h += resid[(size_t)i * HID + c];
  const float a = prelu_a[0];
  out[(size_t)i * HID + c] = (h >= 0.0f) ? h : a * h;
}

// ---------- launch ----------

extern "C" void kernel_launch(void* const* d_in, const int* in_sizes, int n_in,
                              void* d_out, int out_size, void* d_ws, size_t ws_size,
                              hipStream_t stream) {
  const float* x  = (const float*)d_in[0];
  // d_in[1] edge_index: unused by the math
  const float* W1 = (const float*)d_in[2];
  const float* b1 = (const float*)d_in[3];
  const float* W2 = (const float*)d_in[4];
  const float* b2 = (const float*)d_in[5];
  const float* pa = (const float*)d_in[6];

  char* ws = (char*)d_ws;
  _Float16* xh  = (_Float16*)(ws + 0);         //  8,388,608 B (also reused for h1 in f16)
  _Float16* w1h = (_Float16*)(ws + 8388608);   //    131,072 B
  _Float16* w2h = (_Float16*)(ws + 8519680);   //    131,072 B
  float*    sq  = (float*)   (ws + 8650752);   //     65,536 B
  int*      nbr = (int*)     (ws + 8716288);   //    262,144 B
  int*      deg = (int*)     (ws + 8978432);   //     65,536 B
  float*    xt  = (float*)   (ws + 9043968);   // 16,777,216 B
  float*    ef  = (float*)   (ws + 25821184);  // 16,777,216 B
  float*    h1  = (float*)   (ws + 42598400);  // 16,777,216 B  (total ~59.4 MB)

  // --- KNN structure ---
  zero_b32_kernel<<<64, 256, 0, stream>>>((unsigned*)deg, NN);
  f32_to_f16_kernel<<<2048, 256, 0, stream>>>(x, xh, NN * HID);
  f32_to_f16_kernel<<<64, 256, 0, stream>>>(W1, w1h, HID * HID);
  f32_to_f16_kernel<<<64, 256, 0, stream>>>(W2, w2h, HID * HID);
  rownorm_kernel<<<NN, HID, 0, stream>>>(x, sq);
  knn_kernel<<<NN / 16, 32, 0, stream>>>(xh, sq, nbr, deg);

  // --- layer 1: h1 = prelu(hconv(x, W1, b1)) ---
  gemm_kernel<<<NN / 16, 32, 0, stream>>>(xh, w1h, xt);
  zero_b32_kernel<<<4096, 256, 0, stream>>>((unsigned*)ef, NN * HID);
  scatter_kernel<<<NN, HID, 0, stream>>>(xt, nbr, ef);
  gather_kernel<<<NN, HID, 0, stream>>>(ef, deg, nbr, b1, nullptr, pa, h1);

  // --- layer 2: out = prelu(hconv(h1, W2, b2) + x) ---
  f32_to_f16_kernel<<<2048, 256, 0, stream>>>(h1, xh, NN * HID);
  gemm_kernel<<<NN / 16, 32, 0, stream>>>(xh, w2h, xt);
  zero_b32_kernel<<<4096, 256, 0, stream>>>((unsigned*)ef, NN * HID);
  scatter_kernel<<<NN, HID, 0, stream>>>(xt, nbr, ef);
  gather_kernel<<<NN, HID, 0, stream>>>(ef, deg, nbr, b2, x, pa, (float*)d_out);
}
